// ChildSumTreeLSTM_61100204753163
// MI455X (gfx1250) — matrix operand order repaired and checked by reading, not compiled, not run on hardware
//
#include <hip/hip_runtime.h>
#include <hip/hip_bf16.h>
#include <math.h>

// ---------------------------------------------------------------------------
// ChildSum TreeLSTM for MI455X (gfx1250), bf16 WMMA with f32 accumulate.
// Tree structure is static per the reference's _build_tree (LEVEL_COUNTS).
// ---------------------------------------------------------------------------

typedef __attribute__((ext_vector_type(16))) __bf16 v16bf;
typedef __attribute__((ext_vector_type(8)))  float  v8f;

#define HDIM  128
#define HDIM3 384

__device__ __forceinline__ float sig_f(float x) { return 1.0f / (1.0f + __expf(-x)); }

__device__ __forceinline__ v8f wmma_bf16(v16bf a, v16bf b, v8f c) {
    // D = A(16x32 bf16) * B(32x16 bf16) + C(16x16 f32)
    return __builtin_amdgcn_wmma_f32_16x16x32_bf16(
        /*neg_a=*/false, a, /*neg_b=*/false, b,
        /*c_mod=*/(short)0, c, /*reuse_a=*/false, /*reuse_b=*/false);
}

// Load one 16x32 bf16 A fragment from an f32 row (this lane's M row).
// ISA layout: lanes 0-15 hold M=lane, K = {khalf*8..+7} U {16+khalf*8..+7}
// within the 32-wide K chunk; lanes 16-31 are the same rows, other K half.
__device__ __forceinline__ v16bf load_a_frag_f32(const float* __restrict__ row,
                                                 int kc, int khalf) {
    const float4* p0 = reinterpret_cast<const float4*>(row + kc * 32 + khalf * 8);
    const float4* p1 = reinterpret_cast<const float4*>(row + kc * 32 + 16 + khalf * 8);
    float4 x0 = p0[0], x1 = p0[1];
    float4 y0 = p1[0], y1 = p1[1];
    v16bf a;
    a[0]  = (__bf16)x0.x; a[1]  = (__bf16)x0.y; a[2]  = (__bf16)x0.z; a[3]  = (__bf16)x0.w;
    a[4]  = (__bf16)x1.x; a[5]  = (__bf16)x1.y; a[6]  = (__bf16)x1.z; a[7]  = (__bf16)x1.w;
    a[8]  = (__bf16)y0.x; a[9]  = (__bf16)y0.y; a[10] = (__bf16)y0.z; a[11] = (__bf16)y0.w;
    a[12] = (__bf16)y1.x; a[13] = (__bf16)y1.y; a[14] = (__bf16)y1.z; a[15] = (__bf16)y1.w;
    return a;
}

// ---------------------------------------------------------------------------
// Swizzle a row-major f32 weight [128 x N] into B fragments (bf16).
// Fragment (nt, kc): lane holds col = nt*16 + (lane&15),
//                    K = kc*32 + (lane>>4)*8 + j + (j>=8 ? 8 : 0)   (j = 0..15)
// Stored as 16 contiguous bf16 per lane -> two b128 loads at use time.
// ---------------------------------------------------------------------------
__global__ void swizzle_weight_kernel(const float* __restrict__ W,
                                      __bf16* __restrict__ out,
                                      int N, int ntiles) {
    int tid = blockIdx.x * blockDim.x + threadIdx.x;
    int total = ntiles * 4 * 32;
    if (tid >= total) return;
    int lane = tid & 31;
    int fi   = tid >> 5;
    int kc   = fi & 3;
    int nt   = fi >> 2;
    int col  = nt * 16 + (lane & 15);
    int kbase = kc * 32 + (lane >> 4) * 8;
    __bf16* o = out + (size_t)tid * 16;
#pragma unroll
    for (int j = 0; j < 16; ++j)
        o[j] = (__bf16)W[(size_t)(kbase + j + (j >= 8 ? 8 : 0)) * N + col];
}

// ---------------------------------------------------------------------------
// wiou = emb[features] @ W_iou + b_iou ; wf = emb[features] @ W_f + b_f
// One wave per 16-row tile; 32 output ntiles processed in pairs (2
// independent accumulator chains per iteration for WMMA ILP).
// ---------------------------------------------------------------------------
__global__ void input_gemm_kernel(const int* __restrict__ feats,
                                  const float* __restrict__ emb,
                                  const v16bf* __restrict__ bsw_wiou,
                                  const v16bf* __restrict__ bsw_wf,
                                  const float* __restrict__ b_iou,
                                  const float* __restrict__ b_f,
                                  float* __restrict__ wiou,
                                  float* __restrict__ wf,
                                  int M) {
    int lane  = threadIdx.x & 31;
    int wave  = threadIdx.x >> 5;
    int mtile = blockIdx.x * 4 + wave;
    int mtiles = (M + 15) >> 4;
    if (mtile >= mtiles) return;           // wave-uniform: EXEC stays all-ones
    int mrow  = lane & 15;
    int khalf = lane >> 4;
    bool full = (mtile * 16 + 16) <= M;    // wave-uniform

    int row = mtile * 16 + mrow;
    if (row >= M) row = M - 1;
    int feat = feats[row];
    const float* rp = emb + (size_t)feat * HDIM;

    v16bf a[4];
#pragma unroll
    for (int kc = 0; kc < 4; ++kc) a[kc] = load_a_frag_f32(rp, kc, khalf);

    for (int nt = 0; nt < 32; nt += 2) {
        // Pairs never straddle the wiou(24 tiles)/wf(8 tiles) boundary.
        const v16bf* bs0 = (nt < 24) ? (bsw_wiou + (size_t)(nt * 4) * 32)
                                     : (bsw_wf   + (size_t)((nt - 24) * 4) * 32);
        const v16bf* bs1 = bs0 + (size_t)4 * 32;
        v8f acc0 = {0.f, 0.f, 0.f, 0.f, 0.f, 0.f, 0.f, 0.f};
        v8f acc1 = acc0;
#pragma unroll
        for (int kc = 0; kc < 4; ++kc) {
            v16bf b0 = bs0[kc * 32 + lane];
            v16bf b1 = bs1[kc * 32 + lane];
            acc0 = wmma_bf16(a[kc], b0, acc0);
            acc1 = wmma_bf16(a[kc], b1, acc1);
        }

        int    ntl  = (nt < 24) ? nt : (nt - 24);
        int    nc0  = ntl * 16 + mrow;
        float  bi0  = (nt < 24) ? b_iou[nc0] : b_f[nc0];
        float  bi1  = (nt < 24) ? b_iou[nc0 + 16] : b_f[nc0 + 16];
        float* outp = (nt < 24) ? wiou : wf;
        int    ld   = (nt < 24) ? HDIM3 : HDIM;

        if (full) {
#pragma unroll
            for (int r = 0; r < 8; ++r) {   // C layout: VGPR r <-> M = 8*half + r
                int m = mtile * 16 + khalf * 8 + r;
                outp[(size_t)m * ld + nc0]      = acc0[r] + bi0;
                outp[(size_t)m * ld + nc0 + 16] = acc1[r] + bi1;
            }
        } else {
#pragma unroll
            for (int r = 0; r < 8; ++r) {
                int m = mtile * 16 + khalf * 8 + r;
                if (m < M) {
                    outp[(size_t)m * ld + nc0]      = acc0[r] + bi0;
                    outp[(size_t)m * ld + nc0 + 16] = acc1[r] + bi1;
                }
            }
        }
    }
}

// ---------------------------------------------------------------------------
// Level-0 nodes (no children): c = i*u ; h = o*tanh(c)
// ---------------------------------------------------------------------------
__global__ void level0_kernel(const float* __restrict__ wiou,
                              float* __restrict__ h, float* __restrict__ c, int P) {
    int idx = blockIdx.x * blockDim.x + threadIdx.x;
    if (idx >= P * HDIM) return;
    int p = idx >> 7;
    int n = idx & 127;
    const float* wr = wiou + (size_t)p * HDIM3;
    float iv = sig_f(wr[n]);
    float ov = sig_f(wr[HDIM + n]);
    float uv = tanhf(wr[2 * HDIM + n]);
    float nc = iv * uv;
    c[idx] = nc;
    h[idx] = ov * tanhf(nc);
}

__global__ void zero_kernel(float* __restrict__ p, int n) {
    int i = blockIdx.x * blockDim.x + threadIdx.x;
    int stride = gridDim.x * blockDim.x;
    for (; i < n; i += stride) p[i] = 0.0f;
}

// ---------------------------------------------------------------------------
// Edge kernel (level l): fpre = h[child] @ U_f ; f = sigmoid(fpre + wf[parent])
// scatter-add f*c[child] -> c_sum[parent], h[child] -> h_sum[parent].
// 8 ntiles processed in pairs (2 independent WMMA chains).
// ---------------------------------------------------------------------------
__global__ void edge_kernel(const int* __restrict__ adj,
                            const float* __restrict__ h,
                            const float* __restrict__ c,
                            const float* __restrict__ wf,
                            const v16bf* __restrict__ bsw_uf,
                            float* __restrict__ h_sum,
                            float* __restrict__ c_sum,
                            int edge_base, int E, int parent_base) {
    int lane  = threadIdx.x & 31;
    int wave  = threadIdx.x >> 5;
    int mtile = blockIdx.x * 4 + wave;
    int mtiles = (E + 15) >> 4;
    if (mtile >= mtiles) return;
    int mrow  = lane & 15;
    int khalf = lane >> 4;
    bool full = (mtile * 16 + 16) <= E;    // wave-uniform

    int el = mtile * 16 + mrow;
    if (el >= E) el = E - 1;
    int child_row = adj[2 * (edge_base + el) + 1];
    const float* rp = h + (size_t)child_row * HDIM;

    v16bf a[4];
#pragma unroll
    for (int kc = 0; kc < 4; ++kc) a[kc] = load_a_frag_f32(rp, kc, khalf);

    int par[8], chd[8];
#pragma unroll
    for (int r = 0; r < 8; ++r) {
        int m  = mtile * 16 + khalf * 8 + r;
        int mc = (m < E) ? m : (E - 1);
        par[r] = adj[2 * (edge_base + mc)];
        chd[r] = adj[2 * (edge_base + mc) + 1];
    }

    for (int nt = 0; nt < 8; nt += 2) {
        const v16bf* bs0 = bsw_uf + (size_t)(nt * 4) * 32;
        const v16bf* bs1 = bs0 + (size_t)4 * 32;
        v8f acc0 = {0.f, 0.f, 0.f, 0.f, 0.f, 0.f, 0.f, 0.f};
        v8f acc1 = acc0;
#pragma unroll
        for (int kc = 0; kc < 4; ++kc) {
            v16bf b0 = bs0[kc * 32 + lane];
            v16bf b1 = bs1[kc * 32 + lane];
            acc0 = wmma_bf16(a[kc], b0, acc0);
            acc1 = wmma_bf16(a[kc], b1, acc1);
        }

        int nc0 = nt * 16 + mrow;
#pragma unroll
        for (int r = 0; r < 8; ++r) {
            int m = mtile * 16 + khalf * 8 + r;
            if (full || m < E) {
                const float* hr = h + (size_t)chd[r] * HDIM;
                const float* cr = c + (size_t)chd[r] * HDIM;
                const float* wr = wf + (size_t)par[r] * HDIM;
                int po = (par[r] - parent_base) * HDIM;
#pragma unroll
                for (int q = 0; q < 2; ++q) {
                    int   n  = nc0 + q * 16;
                    float av = q ? acc1[r] : acc0[r];
                    float fv = sig_f(av + wr[n]);
                    float fc = fv * cr[n];
                    __hip_atomic_fetch_add(c_sum + po + n, fc,
                                           __ATOMIC_RELAXED, __HIP_MEMORY_SCOPE_AGENT);
                    __hip_atomic_fetch_add(h_sum + po + n, hr[n],
                                           __ATOMIC_RELAXED, __HIP_MEMORY_SCOPE_AGENT);
                }
            }
        }
    }
}

// ---------------------------------------------------------------------------
// Node kernel (level l): iou = wiou[p] + h_sum @ U_iou ; gates ; write h, c.
// Each wave computes gate tiles (j, j+8, j+16) together so i/o/u combine
// in registers (3 independent WMMA chains); A frags loaded once, reused 24x.
// ---------------------------------------------------------------------------
__global__ void node_kernel(const float* __restrict__ h_sum,
                            const float* __restrict__ c_sum,
                            const float* __restrict__ wiou,
                            const v16bf* __restrict__ bsw_uiou,
                            float* __restrict__ h,
                            float* __restrict__ c,
                            int node_base, int P) {
    int lane  = threadIdx.x & 31;
    int wave  = threadIdx.x >> 5;
    int mtile = blockIdx.x * 4 + wave;
    int mtiles = (P + 15) >> 4;
    if (mtile >= mtiles) return;
    int mrow  = lane & 15;
    int khalf = lane >> 4;
    bool full = (mtile * 16 + 16) <= P;    // wave-uniform

    int pl = mtile * 16 + mrow;
    if (pl >= P) pl = P - 1;
    const float* rp = h_sum + (size_t)pl * HDIM;

    v16bf a[4];
#pragma unroll
    for (int kc = 0; kc < 4; ++kc) a[kc] = load_a_frag_f32(rp, kc, khalf);

    for (int j = 0; j < 8; ++j) {
        v8f ai = {0.f, 0.f, 0.f, 0.f, 0.f, 0.f, 0.f, 0.f};
        v8f ao = ai, au = ai;
#pragma unroll
        for (int kc = 0; kc < 4; ++kc) {
            v16bf bi = bsw_uiou[(size_t)((j)      * 4 + kc) * 32 + lane];
            v16bf bo = bsw_uiou[(size_t)((j + 8)  * 4 + kc) * 32 + lane];
            v16bf bu = bsw_uiou[(size_t)((j + 16) * 4 + kc) * 32 + lane];
            ai = wmma_bf16(a[kc], bi, ai);
            ao = wmma_bf16(a[kc], bo, ao);
            au = wmma_bf16(a[kc], bu, au);
        }

        int ncol = j * 16 + mrow;
#pragma unroll
        for (int r = 0; r < 8; ++r) {
            int m = mtile * 16 + khalf * 8 + r;
            if (full || m < P) {
                int p = node_base + m;
                const float* wr = wiou + (size_t)p * HDIM3;
                float iv = sig_f(ai[r] + wr[ncol]);
                float ov = sig_f(ao[r] + wr[HDIM + ncol]);
                float uv = tanhf(au[r] + wr[2 * HDIM + ncol]);
                float nc = iv * uv + c_sum[(size_t)m * HDIM + ncol];
                c[(size_t)p * HDIM + ncol] = nc;
                h[(size_t)p * HDIM + ncol] = ov * tanhf(nc);
            }
        }
    }
}

// ---------------------------------------------------------------------------
// Host-side launch sequence (graph-capture safe: only kernel launches).
// ---------------------------------------------------------------------------
extern "C" void kernel_launch(void* const* d_in, const int* in_sizes, int n_in,
                              void* d_out, int out_size, void* d_ws, size_t ws_size,
                              hipStream_t stream) {
    const int*   feats = (const int*)  d_in[0];
    const int*   adj   = (const int*)  d_in[2];   // [E][2] = {parent, child}
    const float* emb   = (const float*)d_in[4];
    const float* W_iou = (const float*)d_in[5];
    const float* b_iou = (const float*)d_in[6];
    const float* U_iou = (const float*)d_in[7];
    const float* W_f   = (const float*)d_in[8];
    const float* b_f   = (const float*)d_in[9];
    const float* U_f   = (const float*)d_in[10];
    (void)n_in; (void)out_size; (void)ws_size;

    const int N = in_sizes[0];                    // 100000
    float* h = (float*)d_out;
    float* c = (float*)d_out + (size_t)N * HDIM;

    // Static tree structure per reference LEVEL_COUNTS.
    static const int counts[8]   = {60000, 20000, 10000, 5000, 2500, 1500, 800, 200};
    static const int node_off[8] = {0, 60000, 80000, 90000, 95000, 97500, 99000, 99800};

    // Carve workspace (256B-aligned regions).
    char*  ws  = (char*)d_ws;
    size_t off = 0;
    auto carve = [&](size_t bytes) -> char* {
        char* p = ws + off;
        off = (off + bytes + 255) & ~(size_t)255;
        return p;
    };
    float* wiou = (float*)carve((size_t)N * HDIM3 * sizeof(float));   // 153.6 MB
    float* wf   = (float*)carve((size_t)N * HDIM  * sizeof(float));   //  51.2 MB
    float* hsum = (float*)carve((size_t)20000 * HDIM * sizeof(float));
    float* csum = (float*)carve((size_t)20000 * HDIM * sizeof(float));
    v16bf* bsw_wiou = (v16bf*)carve((size_t)24 * 4 * 32 * sizeof(v16bf));
    v16bf* bsw_wf   = (v16bf*)carve((size_t) 8 * 4 * 32 * sizeof(v16bf));
    v16bf* bsw_uiou = (v16bf*)carve((size_t)24 * 4 * 32 * sizeof(v16bf));
    v16bf* bsw_uf   = (v16bf*)carve((size_t) 8 * 4 * 32 * sizeof(v16bf));

    // 1) Swizzle weights to bf16 B-fragments.
    {
        int t24 = 24 * 4 * 32, t8 = 8 * 4 * 32;
        swizzle_weight_kernel<<<(t24 + 127) / 128, 128, 0, stream>>>(W_iou, (__bf16*)bsw_wiou, HDIM3, 24);
        swizzle_weight_kernel<<<(t8  + 127) / 128, 128, 0, stream>>>(W_f,   (__bf16*)bsw_wf,   HDIM,   8);
        swizzle_weight_kernel<<<(t24 + 127) / 128, 128, 0, stream>>>(U_iou, (__bf16*)bsw_uiou, HDIM3, 24);
        swizzle_weight_kernel<<<(t8  + 127) / 128, 128, 0, stream>>>(U_f,   (__bf16*)bsw_uf,   HDIM,   8);
    }

    // 2) Input-side GEMMs for all nodes (fused embedding gather).
    {
        int mtiles = (N + 15) >> 4;
        int blocks = (mtiles + 3) >> 2;            // 4 waves / block
        input_gemm_kernel<<<blocks, 128, 0, stream>>>(feats, emb, bsw_wiou, bsw_wf,
                                                      b_iou, b_f, wiou, wf, N);
    }

    // 3) Level 0 (leaves).
    {
        int P = counts[0];
        level0_kernel<<<(P * HDIM + 255) / 256, 256, 0, stream>>>(wiou, h, c, P);
    }

    // 4) Levels 1..7, bottom-up (stream order gives the dependency).
    for (int l = 1; l < 8; ++l) {
        int E     = counts[l - 1];
        int P     = counts[l];
        int ebase = node_off[l - 1];               // children of level l are level l-1
        int pbase = node_off[l];

        int zn = P * HDIM;
        int zb = (zn + 255) / 256; if (zb > 2048) zb = 2048;
        zero_kernel<<<zb, 256, 0, stream>>>(hsum, zn);
        zero_kernel<<<zb, 256, 0, stream>>>(csum, zn);

        int emt = (E + 15) >> 4;
        edge_kernel<<<(emt + 3) >> 2, 128, 0, stream>>>(adj, h, c, wf, bsw_uf,
                                                        hsum, csum, ebase, E, pbase);

        int pmt = (P + 15) >> 4;
        node_kernel<<<(pmt + 3) >> 2, 128, 0, stream>>>(hsum, csum, wiou, bsw_uiou,
                                                        h, c, pbase, P);
    }
}
